// ScaledDotProductAttention_88476326297955
// MI455X (gfx1250) — matrix operand order repaired and checked by reading, not compiled, not run on hardware
//
#include <hip/hip_runtime.h>
#include <hip/hip_bf16.h>

typedef __attribute__((ext_vector_type(16))) __bf16 v16bf;
typedef __attribute__((ext_vector_type(8)))  float  v8f;

#define B_  2
#define H_  16
#define L_  2048
#define D_  64
#define TQ  16          // q rows per workgroup
#define KC  512         // V chunk (k rows) staged per iteration
#define VTS 520         // padded row stride (bf16 elems) of transposed-V LDS tile

union FragU { int4 q[2]; v16bf v; };

// A-matrix 16-bit fragment: lane holds row M=lane&15;
// elems 0..15 -> K = k0+2*hl8+0..15 contiguous  (hl8 = 8 if lane>=16)
__device__ inline v16bf load_fragA_bf16(const __bf16* row, int k0, int hl8) {
  FragU f;
  f.q[0] = *(const int4*)(row + k0 + 2 * hl8);
  f.q[1] = *(const int4*)(row + k0 + 2 * hl8 + 8);
  return f.v;
}

// B-matrix 16-bit fragment: lane holds col N=lane&15; K = (lane>=16?16:0)+e contiguous.
__device__ inline v16bf load_fragB_bf16(const __bf16* p) {  // p already at k-base
  FragU f;
  f.q[0] = *(const int4*)(p);
  f.q[1] = *(const int4*)(p + 8);
  return f.v;
}

// Same B fragment but sourced from contiguous f32 in global memory (convert to bf16).
__device__ inline v16bf load_fragB_f32(const float* p) {
  v16bf v;
#pragma unroll
  for (int i = 0; i < 16; i += 4) {
    float4 x = *(const float4*)(p + i);
    v[i + 0] = (__bf16)x.x; v[i + 1] = (__bf16)x.y;
    v[i + 2] = (__bf16)x.z; v[i + 3] = (__bf16)x.w;
  }
  return v;
}

__device__ inline v8f wmma_bf16(v16bf a, v16bf b, v8f c) {
  return __builtin_amdgcn_wmma_f32_16x16x32_bf16(false, a, false, b, (short)0, c,
                                                 false, false);
}

__global__ void __launch_bounds__(256)
sdpa_kernel(const float* __restrict__ Q, const float* __restrict__ K,
            const float* __restrict__ V, float* __restrict__ ctx,
            float* __restrict__ attn) {
  __shared__ __attribute__((aligned(16))) float  sS[TQ * L_];   // 128 KB scores; reused: V^T tile + reduce buf
  __shared__ __attribute__((aligned(16))) __bf16 sP[TQ * L_];   // 64 KB probs (bf16)
  __shared__ __attribute__((aligned(16))) __bf16 sQ[TQ * D_];   // 2 KB Q*scale (bf16)

  const int tid  = threadIdx.x;
  const int lane = tid & 31;
  const int w    = tid >> 5;                 // 8 waves
  const int hl8  = (lane >= 16) ? 8 : 0;
  const int hl16 = (lane >= 16) ? 16 : 0;
  const int n    = lane & 15;

  const int bid  = blockIdx.x;
  const int qblk = bid & (L_ / TQ - 1);      // 128 q-blocks per (b,h)
  const int bh   = bid >> 7;
  const int b    = bh >> 4, h = bh & 15;
  const int qbase = qblk * TQ;
  const float scale = 0.125f;                // D^-0.5

  const float* Qp = Q + (size_t)bh * L_ * D_;
  const float* Kp = K + (size_t)bh * L_ * D_;
  const float* Vp = V + (size_t)bh * L_ * D_;
  float* attnBase = attn + ((size_t)bh * L_ + qbase) * L_;

  // ---- Phase 0: stage Q*scale as bf16 ----
  for (int i = tid; i < TQ * D_; i += 256) {
    int r = i >> 6, d = i & 63;
    sQ[r * D_ + d] = (__bf16)(Qp[(size_t)(qbase + r) * D_ + d] * scale);
  }
  __syncthreads();

  const __bf16* qrow = sQ + (size_t)n * D_;
  v16bf aq0 = load_fragA_bf16(qrow, 0,  hl8);
  v16bf aq1 = load_fragA_bf16(qrow, 32, hl8);

  // ---- Phase 1: S = (Q*scale) @ K^T, one 16-row strip in LDS ----
  for (int t = w; t < L_ / 16; t += 8) {
    const float* krow = Kp + (size_t)(t * 16 + n) * D_ + hl16;
    v8f c = {};
    c = wmma_bf16(aq0, load_fragB_f32(krow),      c);
    c = wmma_bf16(aq1, load_fragB_f32(krow + 32), c);
#pragma unroll
    for (int r = 0; r < 8; r++)
      sS[(size_t)(r + hl8) * L_ + t * 16 + n] = c[r];
  }
  __syncthreads();

  // ---- Phase 2: softmax per row; attn_out = 0.5 * softmax (temq term is identically 1.0) ----
#pragma unroll
  for (int rr = 0; rr < 2; rr++) {
    const int row = w + rr * 8;
    float* Srow = sS + (size_t)row * L_;
    float m = -3.402823466e38f;
    for (int j = lane; j < L_; j += 32) m = fmaxf(m, Srow[j]);
#pragma unroll
    for (int s = 16; s >= 1; s >>= 1) m = fmaxf(m, __shfl_xor(m, s, 32));
    float sum = 0.0f;
    for (int j = lane; j < L_; j += 32) {
      float e = __expf(Srow[j] - m);
      Srow[j] = e;
      sum += e;
    }
#pragma unroll
    for (int s = 16; s >= 1; s >>= 1) sum += __shfl_xor(sum, s, 32);
    const float inv = 0.5f / sum;
    float*  arow = attnBase + (size_t)row * L_;
    __bf16* prow = sP + (size_t)row * L_;
    for (int j = lane; j < L_; j += 32) {
      float p = Srow[j] * inv;
      __builtin_nontemporal_store(p, arow + j);   // 536 MB stream: keep out of caches
      prow[j] = (__bf16)p;
    }
  }
  __syncthreads();

  // ---- Phase 3: context = P @ V (P already carries the 0.5 factor) ----
  v8f cacc = {};
  const int dtile = w & 3;        // which 16-wide d tile of D=64
  const int khalf = w >> 2;       // which half of each k chunk
  __bf16* vt = (__bf16*)sS;       // V^T tile, [D_][VTS] padded
  const __bf16* prow = sP + (size_t)n * L_;

  for (int c = 0; c < L_ / KC; c++) {
    __syncthreads();
    // stage V chunk transposed -> bf16
    for (int i = tid; i < KC * D_ / 4; i += 256) {
      const int e = i * 4;
      const int k = e >> 6, d = e & 63;
      float4 x = *(const float4*)(Vp + (size_t)(c * KC + k) * D_ + d);
      vt[(size_t)(d + 0) * VTS + k] = (__bf16)x.x;
      vt[(size_t)(d + 1) * VTS + k] = (__bf16)x.y;
      vt[(size_t)(d + 2) * VTS + k] = (__bf16)x.z;
      vt[(size_t)(d + 3) * VTS + k] = (__bf16)x.w;
    }
    __syncthreads();
#pragma unroll
    for (int s = 0; s < KC / 64; s++) {
      const int kl = khalf * (KC / 2) + s * 32;     // local k in chunk
      const int kk = c * KC + kl;                   // global k for P
      v16bf a  = load_fragA_bf16(prow, kk, hl8);
      v16bf bv = load_fragB_bf16(vt + (size_t)(dtile * 16 + n) * VTS + kl + hl16);
      cacc = wmma_bf16(a, bv, cacc);
    }
  }

  // cross-wave (k-half) reduction via LDS region past the V^T tile
  float* red = sS + 16640;                          // byte 66560, past vt (64*520*2 B)
  if (w >= 4) {
#pragma unroll
    for (int r = 0; r < 8; r++)
      red[((w - 4) * 8 + r) * 32 + lane] = cacc[r];
  }
  __syncthreads();
  if (w < 4) {
#pragma unroll
    for (int r = 0; r < 8; r++)
      cacc[r] += red[(w * 8 + r) * 32 + lane];
    // ctx layout [B, Lq, H, D]
#pragma unroll
    for (int r = 0; r < 8; r++) {
      const int qrow2 = qbase + r + hl8;
      const int d     = dtile * 16 + n;
      ctx[(((size_t)b * L_ + qrow2) * H_ + h) * D_ + d] = cacc[r];
    }
  }
}

extern "C" void kernel_launch(void* const* d_in, const int* in_sizes, int n_in,
                              void* d_out, int out_size, void* d_ws, size_t ws_size,
                              hipStream_t stream) {
  const float* Q = (const float*)d_in[0];
  const float* K = (const float*)d_in[1];
  const float* V = (const float*)d_in[2];
  // W1,b1,W2,b2 (d_in[3..6]) are mathematically dead: softmax over a singleton
  // axis makes temq == 1.0, so attn is just 0.5 * softmax(QK^T * scale).
  float* ctx  = (float*)d_out;                                // [B,L,H,D]
  float* attn = (float*)d_out + (size_t)B_ * L_ * H_ * D_;    // [B,H,L,L]
  sdpa_kernel<<<dim3(B_ * H_ * (L_ / TQ)), dim3(256), 0, stream>>>(Q, K, V, ctx, attn);
}
